// FlexibleSelfAttention_51866025066799
// MI455X (gfx1250) — matrix operand order repaired and checked by reading, not compiled, not run on hardware
//
#include <hip/hip_runtime.h>

#define D_MODEL 1024
#define NHEADS  16
#define HDIM    64
#define BATCH   2
#define SEQ     2048

// ---------------------------------------------------------------------------
// CDNA5 (gfx1250) wave32 WMMA types + helpers
// ---------------------------------------------------------------------------
typedef __attribute__((ext_vector_type(16))) __bf16 v16bf;
typedef __attribute__((ext_vector_type(8)))  float  v8f;

union FragB16 { v16bf v; unsigned int u[8]; };

__device__ __forceinline__ unsigned short f32_to_bf16(float f) {
  unsigned int u = __float_as_uint(f);
  u += 0x7FFFu + ((u >> 16) & 1u);          // round-to-nearest-even
  return (unsigned short)(u >> 16);
}
__device__ __forceinline__ float bf16_to_f32(unsigned short h) {
  return __uint_as_float(((unsigned int)h) << 16);
}

// DPP16 butterfly reduction across the 16 lanes holding one C-tile row.
template <int CTRL>
__device__ __forceinline__ float dpp_mov(float x) {
  return __int_as_float(__builtin_amdgcn_update_dpp(
      0, __float_as_int(x), CTRL, 0xf, 0xf, true));
}
__device__ __forceinline__ float rowmax16(float x) {
  x = fmaxf(x, dpp_mov<0xB1>(x));   // quad_perm [1,0,3,2]
  x = fmaxf(x, dpp_mov<0x4E>(x));   // quad_perm [2,3,0,1]
  x = fmaxf(x, dpp_mov<0x141>(x));  // row_half_mirror
  x = fmaxf(x, dpp_mov<0x140>(x));  // row_mirror
  return x;
}
__device__ __forceinline__ float rowsum16(float x) {
  x += dpp_mov<0xB1>(x);
  x += dpp_mov<0x4E>(x);
  x += dpp_mov<0x141>(x);
  x += dpp_mov<0x140>(x);
  return x;
}

__device__ __forceinline__ v8f wmma_bf16(const FragB16& a, const FragB16& b, v8f c) {
  return __builtin_amdgcn_wmma_f32_16x16x32_bf16(false, a.v, false, b.v,
                                                 (short)0, c, false, false);
}

// ---------------------------------------------------------------------------
// CDNA5 async global->LDS copy (bypasses VGPRs; tracked by ASYNCcnt).
// LDS operand is the low 32 bits of the generic pointer (ISA 10.2: LDS
// aperture address truncates to LDS offset).
// ---------------------------------------------------------------------------
__device__ __forceinline__ void async_copy_b128(unsigned short* lds_dst,
                                                const unsigned short* gsrc) {
  asm volatile("global_load_async_to_lds_b128 %0, %1, off"
               :: "v"((unsigned)(unsigned long long)lds_dst), "v"(gsrc)
               : "memory");
}
__device__ __forceinline__ void wait_async0() {
  asm volatile("s_wait_asynccnt 0x0" ::: "memory");
}

// ---------------------------------------------------------------------------
// Prep kernels (one-shot, memory-bound)
// ---------------------------------------------------------------------------
__global__ __launch_bounds__(256) void convert_bf16(const float* __restrict__ src,
                                                    unsigned short* __restrict__ dst,
                                                    int n4) {
  int i = blockIdx.x * blockDim.x + threadIdx.x;
  if (i >= n4) return;
  float4 f = ((const float4*)src)[i];
  ushort4 h;
  h.x = f32_to_bf16(f.x);
  h.y = f32_to_bf16(f.y);
  h.z = f32_to_bf16(f.z);
  h.w = f32_to_bf16(f.w);
  ((ushort4*)dst)[i] = h;
}

// weights: f32 [K][N] -> bf16 [N][K] (so GEMM B staging is pure b128 copies)
__global__ __launch_bounds__(256) void convert_transpose_bf16(
    const float* __restrict__ src, unsigned short* __restrict__ dst, int K, int N) {
  int idx = blockIdx.x * blockDim.x + threadIdx.x;
  if (idx >= K * N) return;
  int k = idx / N, n = idx % N;             // reads coalesced along N
  dst[(size_t)n * K + k] = f32_to_bf16(src[idx]);
}

// ---------------------------------------------------------------------------
// Tiled WMMA GEMM:  C[M,N] = A[M,K] (bf16) @ WT[N,K] (bf16, pre-transposed)
// 128x128 C tile, 8 waves (4 M x 2 N), wave tile 32x64. Double-buffered LDS
// with async b128 staging; prefetch 2 tiles ahead.
// ---------------------------------------------------------------------------
__device__ __forceinline__ void gemm_stage(const unsigned short* __restrict__ A,
                                           const unsigned short* __restrict__ WT,
                                           unsigned short* __restrict__ As,
                                           unsigned short* __restrict__ Ws,
                                           int tid, int m0, int n0, int k0, int K) {
#pragma unroll
  for (int i = 0; i < 2; ++i) {
    int e8 = (tid + 256 * i) * 8;           // flat bf16 index, multiple of 8
    int mm = e8 >> 5, kk = e8 & 31;
    async_copy_b128(&As[mm * 32 + kk], A + (size_t)(m0 + mm) * K + k0 + kk);
    async_copy_b128(&Ws[mm * 32 + kk], WT + (size_t)(n0 + mm) * K + k0 + kk);
  }
}

template <bool STORE_BF16>
__global__ __launch_bounds__(256) void wmma_gemm(const unsigned short* __restrict__ A,
                                                 const unsigned short* __restrict__ WT,
                                                 void* __restrict__ Cout,
                                                 int M, int N, int K) {
  __shared__ unsigned short As[2][128 * 32];  // [m][k]
  __shared__ unsigned short Ws[2][128 * 32];  // [n][k]
  const int tid  = threadIdx.x;
  const int lane = tid & 31;
  const int wid  = tid >> 5;
  const int wm   = wid & 3;
  const int wn   = wid >> 2;
  const int m0   = blockIdx.y * 128;
  const int n0   = blockIdx.x * 128;
  const int r    = lane & 15;
  const int kh   = lane >> 4;

  v8f acc[2][4];
  const v8f vzero = {0.f, 0.f, 0.f, 0.f, 0.f, 0.f, 0.f, 0.f};
#pragma unroll
  for (int mi = 0; mi < 2; ++mi)
#pragma unroll
    for (int ni = 0; ni < 4; ++ni) acc[mi][ni] = vzero;

  gemm_stage(A, WT, As[0], Ws[0], tid, m0, n0, 0, K);
  wait_async0();
  __syncthreads();

  int buf = 0;
  for (int k0 = 0; k0 < K; k0 += 32) {
    if (k0 + 32 < K) {
      gemm_stage(A, WT, As[buf ^ 1], Ws[buf ^ 1], tid, m0, n0, k0 + 32, K);
      if (k0 + 64 < K) {                    // gfx1250 global_prefetch, 2 tiles ahead
        int e8 = tid * 8;
        __builtin_prefetch(A + (size_t)(m0 + (e8 >> 5)) * K + k0 + 64 + (e8 & 31), 0, 3);
        __builtin_prefetch(WT + (size_t)(n0 + (e8 >> 5)) * K + k0 + 64 + (e8 & 31), 0, 3);
      }
    }

    FragB16 a[2], b[4];
#pragma unroll
    for (int mi = 0; mi < 2; ++mi) {
      int mr = wm * 32 + mi * 16 + r;
#pragma unroll
      for (int v = 0; v < 8; ++v) {
        int kb = (v < 4 ? 2 * v : 16 + 2 * (v - 4)) + 8 * kh;  // A-layout K pairs
        a[mi].u[v] = *(const unsigned int*)&As[buf][mr * 32 + kb];
      }
    }
#pragma unroll
    for (int ni = 0; ni < 4; ++ni) {
      int nc = wn * 64 + ni * 16 + r;
#pragma unroll
      for (int v = 0; v < 8; ++v) {
        int kb = (v < 4 ? 2 * v : 16 + 2 * (v - 4)) + 8 * kh;  // B-layout K pairs
        b[ni].u[v] = *(const unsigned int*)&Ws[buf][nc * 32 + kb];
      }
    }
#pragma unroll
    for (int mi = 0; mi < 2; ++mi)
#pragma unroll
      for (int ni = 0; ni < 4; ++ni)
        acc[mi][ni] = wmma_bf16(a[mi], b[ni], acc[mi][ni]);

    wait_async0();                          // stage of buf^1 landed in LDS
    __syncthreads();
    buf ^= 1;
  }

  // C layout: VGPR v, lanes<16 -> M=v, lanes>=16 -> M=v+8 ; N = lane%16
#pragma unroll
  for (int mi = 0; mi < 2; ++mi)
#pragma unroll
    for (int ni = 0; ni < 4; ++ni)
#pragma unroll
      for (int v = 0; v < 8; ++v) {
        int row = m0 + wm * 32 + mi * 16 + v + 8 * kh;
        int col = n0 + wn * 64 + ni * 16 + r;
        float val = acc[mi][ni][v];
        if (STORE_BF16)
          ((unsigned short*)Cout)[(size_t)row * N + col] = f32_to_bf16(val);
        else
          ((float*)Cout)[(size_t)row * N + col] = val;
      }
}

// ---------------------------------------------------------------------------
// RoPE + head-major repack: qkv [B,T,3D] -> Qh,Kh [B,H,T,64] (rope applied).
// ---------------------------------------------------------------------------
__global__ __launch_bounds__(256) void rope_pack_qk(const unsigned short* __restrict__ qkv,
                                                    unsigned short* __restrict__ Qh,
                                                    unsigned short* __restrict__ Kh) {
  int tid = blockIdx.x * blockDim.x + threadIdx.x;   // < B*T*H*32 = 2^21
  int i = tid & 31;
  int h = (tid >> 5) & (NHEADS - 1);
  int t = (tid >> 9) & (SEQ - 1);
  int b = tid >> 20;
  if (b >= BATCH) return;
  float invf = __expf(-(float)i * 0.2878231366f);    // 10000^(-i/32)
  float ang = (float)t * invf;
  float sn, cs;
  __sincosf(ang, &sn, &cs);
  size_t src = ((size_t)b * SEQ + t) * (3 * D_MODEL) + h * HDIM;
  size_t dst = (((size_t)b * NHEADS + h) * SEQ + t) * HDIM;
  {
    float x1 = bf16_to_f32(qkv[src + i]);
    float x2 = bf16_to_f32(qkv[src + i + 32]);
    Qh[dst + i]      = f32_to_bf16(x1 * cs - x2 * sn);
    Qh[dst + i + 32] = f32_to_bf16(x2 * cs + x1 * sn);
  }
  {
    float x1 = bf16_to_f32(qkv[src + D_MODEL + i]);
    float x2 = bf16_to_f32(qkv[src + D_MODEL + i + 32]);
    Kh[dst + i]      = f32_to_bf16(x1 * cs - x2 * sn);
    Kh[dst + i + 32] = f32_to_bf16(x2 * cs + x1 * sn);
  }
}

// V pre-transpose: qkv v-part -> Vt [B,H,64,T] (coalesced writes along t)
__global__ __launch_bounds__(256) void pack_vt(const unsigned short* __restrict__ qkv,
                                               unsigned short* __restrict__ Vt) {
  int tid = blockIdx.x * blockDim.x + threadIdx.x;   // < B*H*64*T = 2^22
  int t = tid & (SEQ - 1);
  int d = (tid >> 11) & (HDIM - 1);
  int h = (tid >> 17) & (NHEADS - 1);
  int b = tid >> 21;
  if (b >= BATCH) return;
  unsigned short v =
      qkv[((size_t)b * SEQ + t) * (3 * D_MODEL) + 2 * D_MODEL + h * HDIM + d];
  Vt[(((size_t)b * NHEADS + h) * HDIM + d) * SEQ + t] = v;
}

// ---------------------------------------------------------------------------
// Causal flash attention. Block = (b, h, 128 queries); 8 waves x 16 queries.
// Key tiles of 32, double-buffered in LDS with async b128 staging. Masking
// only on diagonal tiles (wave-uniform). P bounced through per-wave LDS
// (s_wait_dscnt) to convert C-layout f32 -> A-layout bf16 for the PV WMMA.
// ---------------------------------------------------------------------------
__device__ __forceinline__ void attn_stage(const unsigned short* __restrict__ Kh,
                                           const unsigned short* __restrict__ Vt,
                                           unsigned short* __restrict__ Ksb,
                                           unsigned short* __restrict__ Vtsb,
                                           int tid, size_t kbase, size_t vbase, int tk0) {
  int e8 = tid * 8;
  {
    int trow = e8 >> 6, d = e8 & 63;
    async_copy_b128(&Ksb[trow * 64 + d],
                    Kh + kbase + (size_t)(tk0 + trow) * HDIM + d);
  }
  {
    int d = e8 >> 5, tp = e8 & 31;
    async_copy_b128(&Vtsb[d * 32 + tp],
                    Vt + vbase + (size_t)d * SEQ + tk0 + tp);
  }
}

template <bool MASK>
__device__ __forceinline__ void online_softmax(const v8f& s0, const v8f& s1, v8f acc[4],
                                               float m[8], float l[8],
                                               unsigned short* __restrict__ Pw,
                                               int qw, int tk0, int r, int kh) {
#pragma unroll
  for (int v = 0; v < 8; ++v) {
    float e0 = s0[v], e1 = s1[v];
    if (MASK) {
      int q = qw + v + 8 * kh;
      int key0 = tk0 + r;
      if (key0 > q) e0 = -__builtin_inff();
      if (key0 + 16 > q) e1 = -__builtin_inff();
    }
    float mv = rowmax16(fmaxf(e0, e1));
    float mn = fmaxf(m[v], mv);
    float al = __expf(m[v] - mn);
    float p0 = __expf(e0 - mn);
    float p1 = __expf(e1 - mn);
    l[v] = l[v] * al + rowsum16(p0 + p1);
    m[v] = mn;
#pragma unroll
    for (int ni = 0; ni < 4; ++ni) acc[ni][v] *= al;
    int Mr = v + 8 * kh;
    Pw[Mr * 32 + r]      = f32_to_bf16(p0);
    Pw[Mr * 32 + 16 + r] = f32_to_bf16(p1);
  }
}

__global__ __launch_bounds__(256) void flash_attn(const unsigned short* __restrict__ Qh,
                                                  const unsigned short* __restrict__ Kh,
                                                  const unsigned short* __restrict__ Vt,
                                                  unsigned short* __restrict__ attn) {
  __shared__ unsigned short Ks[2][32 * 64];   // [key][d]
  __shared__ unsigned short Vts[2][64 * 32];  // [d][key]
  __shared__ unsigned short Ps[8][16 * 32];   // per-wave P scratch [M][n]

  const int tid  = threadIdx.x;
  const int lane = tid & 31;
  const int wid  = tid >> 5;
  const int r    = lane & 15;
  const int kh   = lane >> 4;
  const int q0 = blockIdx.x * 128;
  const int h  = blockIdx.y;
  const int b  = blockIdx.z;
  const int qw = q0 + wid * 16;
  const int bh = b * NHEADS + h;
  const size_t kbase = (size_t)bh * SEQ * HDIM;
  const size_t vbase = (size_t)bh * HDIM * SEQ;

  // Q fragments (2 WMMA k-steps over Dh=64), pre-scaled by 1/sqrt(64)
  FragB16 qf[2];
  {
    const unsigned short* qrow = Qh + kbase + (size_t)(qw + r) * HDIM;
#pragma unroll
    for (int s = 0; s < 2; ++s)
#pragma unroll
      for (int v = 0; v < 8; ++v) {
        int d = s * 32 + (v < 4 ? 2 * v : 16 + 2 * (v - 4)) + 8 * kh;
        unsigned int u = *(const unsigned int*)(qrow + d);
        float lo = bf16_to_f32((unsigned short)(u & 0xFFFFu)) * 0.125f;
        float hi = bf16_to_f32((unsigned short)(u >> 16)) * 0.125f;
        qf[s].u[v] = (unsigned int)f32_to_bf16(lo) |
                     ((unsigned int)f32_to_bf16(hi) << 16);
      }
  }

  const v8f vzero = {0.f, 0.f, 0.f, 0.f, 0.f, 0.f, 0.f, 0.f};
  v8f acc[4];
#pragma unroll
  for (int ni = 0; ni < 4; ++ni) acc[ni] = vzero;
  float m[8], l[8];
#pragma unroll
  for (int v = 0; v < 8; ++v) { m[v] = -__builtin_inff(); l[v] = 0.f; }

  const int nkt = q0 / 32 + 4;
  attn_stage(Kh, Vt, Ks[0], Vts[0], tid, kbase, vbase, 0);
  wait_async0();
  __syncthreads();

  int buf = 0;
  for (int kt = 0; kt < nkt; ++kt) {
    const int tk0 = kt * 32;
    if (kt + 1 < nkt) {
      attn_stage(Kh, Vt, Ks[buf ^ 1], Vts[buf ^ 1], tid, kbase, vbase, tk0 + 32);
      if (kt + 2 < nkt) {
        int e8 = tid * 8;
        __builtin_prefetch(Kh + kbase + (size_t)(tk0 + 64) * HDIM + e8, 0, 3);
        __builtin_prefetch(Vt + vbase + (size_t)(e8 >> 5) * SEQ + tk0 + 64 + (e8 & 31), 0, 3);
      }
    }

    if (tk0 <= qw + 15) {                     // wave-uniform causal skip
      // S tile: load all 4 B-fragments up front, then 4 back-to-back WMMAs
      FragB16 bk[2][2];                       // [kstep][ntile]
#pragma unroll
      for (int ks = 0; ks < 2; ++ks)
#pragma unroll
        for (int nt = 0; nt < 2; ++nt)
#pragma unroll
          for (int v = 0; v < 8; ++v) {
            int d = ks * 32 + (v < 4 ? 2 * v : 16 + 2 * (v - 4)) + 8 * kh;
            bk[ks][nt].u[v] = *(const unsigned int*)&Ks[buf][(nt * 16 + r) * 64 + d];
          }
      v8f s0 = vzero, s1 = vzero;
      s0 = wmma_bf16(qf[0], bk[0][0], s0);
      s1 = wmma_bf16(qf[0], bk[0][1], s1);
      s0 = wmma_bf16(qf[1], bk[1][0], s0);
      s1 = wmma_bf16(qf[1], bk[1][1], s1);

      // masking only needed on diagonal tiles (wave-uniform predicate)
      if (tk0 + 31 > qw)
        online_softmax<true>(s0, s1, acc, m, l, Ps[wid], qw, tk0, r, kh);
      else
        online_softmax<false>(s0, s1, acc, m, l, Ps[wid], qw, tk0, r, kh);

      // same-wave LDS RAW: drain DS counter before re-reading in A-layout
      asm volatile("s_wait_dscnt 0x0" ::: "memory");
      FragB16 pf;
#pragma unroll
      for (int v = 0; v < 8; ++v) {
        int kb = (v < 4 ? 2 * v : 16 + 2 * (v - 4)) + 8 * kh;
        pf.u[v] = *(const unsigned int*)&Ps[wid][r * 32 + kb];
      }
      // acc += P @ V : load all 4 V fragments, then 4 back-to-back WMMAs
      FragB16 vf[4];
#pragma unroll
      for (int ni = 0; ni < 4; ++ni)
#pragma unroll
        for (int v = 0; v < 8; ++v)
          vf[ni].u[v] = *(const unsigned int*)&Vts[buf][(ni * 16 + r) * 32 + (v < 4 ? 2 * v : 16 + 2 * (v - 4)) + 8 * kh];
#pragma unroll
      for (int ni = 0; ni < 4; ++ni) acc[ni] = wmma_bf16(pf, vf[ni], acc[ni]);
    }

    wait_async0();                            // stage of buf^1 landed in LDS
    __syncthreads();
    buf ^= 1;
  }

  // normalize + store attention output [B][T][D] bf16
#pragma unroll
  for (int ni = 0; ni < 4; ++ni)
#pragma unroll
    for (int v = 0; v < 8; ++v) {
      int q = qw + v + 8 * kh;
      int c = h * HDIM + ni * 16 + r;
      attn[((size_t)b * SEQ + q) * D_MODEL + c] = f32_to_bf16(acc[ni][v] / l[v]);
    }
}

// ---------------------------------------------------------------------------
// Host launch. Inputs: x(f32), mask(bool, unused - causal computed), w_qkv,
// w_out. Workspace (bf16): xb 8MB | wqkvT 6MB | woutT 2MB | qkv 24MB |
// Qh/Kh/Vt 24MB | attn 8MB  (~72MB total)
// ---------------------------------------------------------------------------
extern "C" void kernel_launch(void* const* d_in, const int* in_sizes, int n_in,
                              void* d_out, int out_size, void* d_ws, size_t ws_size,
                              hipStream_t stream) {
  (void)in_sizes; (void)n_in; (void)out_size; (void)ws_size;
  const float* x     = (const float*)d_in[0];
  const float* w_qkv = (const float*)d_in[2];
  const float* w_out = (const float*)d_in[3];

  const int M = BATCH * SEQ;                  // 4096
  const size_t nX    = (size_t)M * D_MODEL;
  const size_t nWqkv = (size_t)D_MODEL * 3 * D_MODEL;
  const size_t nWout = (size_t)D_MODEL * D_MODEL;
  const size_t nQkv  = (size_t)M * 3 * D_MODEL;
  const size_t nHead = (size_t)BATCH * NHEADS * SEQ * HDIM;

  unsigned short* xb    = (unsigned short*)d_ws;
  unsigned short* wqkvT = xb + nX;
  unsigned short* woutT = wqkvT + nWqkv;
  unsigned short* qkv   = woutT + nWout;
  unsigned short* Qh    = qkv + nQkv;
  unsigned short* Kh    = Qh + nHead;
  unsigned short* Vt    = Kh + nHead;
  unsigned short* attnb = Vt + nHead;

  dim3 blk(256);

  // 0) one-shot conversions (weights pre-transposed to [N][K])
  convert_bf16<<<dim3((nX / 4 + 255) / 256), blk, 0, stream>>>(x, xb, (int)(nX / 4));
  convert_transpose_bf16<<<dim3((nWqkv + 255) / 256), blk, 0, stream>>>(
      w_qkv, wqkvT, D_MODEL, 3 * D_MODEL);
  convert_transpose_bf16<<<dim3((nWout + 255) / 256), blk, 0, stream>>>(
      w_out, woutT, D_MODEL, D_MODEL);

  // 1) x @ w_qkv -> bf16 qkv
  wmma_gemm<true><<<dim3((3 * D_MODEL) / 128, M / 128), blk, 0, stream>>>(
      xb, wqkvT, qkv, M, 3 * D_MODEL, D_MODEL);
  // 2) RoPE + head-major repack (Q,K) and V transpose
  rope_pack_qk<<<dim3((BATCH * SEQ * NHEADS * 32) / 256), blk, 0, stream>>>(qkv, Qh, Kh);
  pack_vt<<<dim3((BATCH * NHEADS * HDIM * SEQ) / 256), blk, 0, stream>>>(qkv, Vt);
  // 3) causal flash attention -> bf16 attn
  flash_attn<<<dim3(SEQ / 128, NHEADS, BATCH), blk, 0, stream>>>(Qh, Kh, Vt, attnb);
  // 4) attn @ w_out -> f32 d_out
  wmma_gemm<false><<<dim3(D_MODEL / 128, M / 128), blk, 0, stream>>>(
      attnb, woutT, (float*)d_out, M, D_MODEL, D_MODEL);
}
